// RecommenderModel_78262894068429
// MI455X (gfx1250) — matrix operand order repaired and checked by reading, not compiled, not run on hardware
//
#include <hip/hip_runtime.h>
#include <hip/hip_bf16.h>

// ---------------------------------------------------------------------------
// Types for CDNA5 WMMA (wave32): v_wmma_f32_16x16x32_bf16
// ---------------------------------------------------------------------------
typedef __attribute__((ext_vector_type(16))) __bf16         v16bf;
typedef __attribute__((ext_vector_type(16))) unsigned short v16u;
typedef __attribute__((ext_vector_type(8)))  unsigned short v8u;
typedef __attribute__((ext_vector_type(8)))  float          v8f;

#define N_USERS 4096
#define N_ITEMS 4096
#define NN      8192   // N_USERS + N_ITEMS
#define D       64
#define BB      4096

// gfx1250 async global->LDS copy (ASYNCcnt) if the toolchain exposes it.
#if defined(__AMDGCN__) && __has_builtin(__builtin_amdgcn_global_load_async_to_lds_b128)
#define USE_ASYNC_LDS 1
#else
#define USE_ASYNC_LDS 0
#endif

#if defined(__AMDGCN__) && __has_builtin(__builtin_amdgcn_s_barrier_signal) && \
    __has_builtin(__builtin_amdgcn_s_barrier_wait)
#define USE_SPLIT_BARRIER 1
#else
#define USE_SPLIT_BARRIER 0
#endif

__device__ __forceinline__ unsigned short f32_to_bf16(float f) {
    unsigned int u = __float_as_uint(f);
    return (unsigned short)((u + 0x8000u) >> 16);   // round-to-nearest-ish
}

__device__ __forceinline__ v8f wmma_bf16(v16bf a, v16bf b, v8f c) {
    return __builtin_amdgcn_wmma_f32_16x16x32_bf16(
        /*neg_a=*/false, a, /*neg_b=*/false, b,
        /*c_mod=*/(short)0, c, /*reuse_a=*/false, /*reuse_b=*/false);
}

// A-operand (16-bit A 16x32 layout): per lane two contiguous 8-elem K chunks
// at +0 and +16 from (row, ktile*32 + half*8).
__device__ __forceinline__ v16bf load_a_f32src(const float* __restrict__ p) {
    float4 x0 = *(const float4*)(p);
    float4 x1 = *(const float4*)(p + 4);
    float4 x2 = *(const float4*)(p + 16);
    float4 x3 = *(const float4*)(p + 20);
    v16u u;
    u[0]=f32_to_bf16(x0.x);  u[1]=f32_to_bf16(x0.y);  u[2]=f32_to_bf16(x0.z);  u[3]=f32_to_bf16(x0.w);
    u[4]=f32_to_bf16(x1.x);  u[5]=f32_to_bf16(x1.y);  u[6]=f32_to_bf16(x1.z);  u[7]=f32_to_bf16(x1.w);
    u[8]=f32_to_bf16(x2.x);  u[9]=f32_to_bf16(x2.y);  u[10]=f32_to_bf16(x2.z); u[11]=f32_to_bf16(x2.w);
    u[12]=f32_to_bf16(x3.x); u[13]=f32_to_bf16(x3.y); u[14]=f32_to_bf16(x3.z); u[15]=f32_to_bf16(x3.w);
    return __builtin_bit_cast(v16bf, u);
}

struct bfpair { v8u lo, hi; };
__device__ __forceinline__ v16bf load_a_bf16src(const unsigned short* __restrict__ p) {
    bfpair t;
    t.lo = *(const v8u*)(p);         // K chunk c*8 .. c*8+7
    t.hi = *(const v8u*)(p + 16);    // K chunk c*8+16 .. c*8+23
    return __builtin_bit_cast(v16bf, t);
}

// ---- cooperative B-tile staging: 4 KB tile, 256 threads x 16 bytes --------
#if USE_ASYNC_LDS
typedef int v4i_gcc __attribute__((vector_size(4 * sizeof(int))));
#endif
__device__ __forceinline__ void stage_b16B(const unsigned short* __restrict__ g,
                                           unsigned short* l) {
#if USE_ASYNC_LDS
    __builtin_amdgcn_global_load_async_to_lds_b128(
        (__attribute__((address_space(1))) v4i_gcc*)g,
        (__attribute__((address_space(3))) v4i_gcc*)l,
        0, 0);
#else
    *(v8u*)l = *(const v8u*)g;
#endif
}

// Barrier that makes the staged LDS tile visible WITHOUT draining the global
// loadcnt (wave-private A prefetch must stay in flight across it).
// __syncthreads() would emit s_wait_loadcnt 0 and expose full HBM latency.
__device__ __forceinline__ void stage_fence() {
#if USE_ASYNC_LDS && USE_SPLIT_BARRIER
#if __has_builtin(__builtin_amdgcn_s_wait_asynccnt)
    __builtin_amdgcn_s_wait_asynccnt(0);
#else
    asm volatile("s_wait_asynccnt 0" ::: "memory");
#endif
    asm volatile("" ::: "memory");                 // no IR reordering across barrier
    __builtin_amdgcn_s_barrier_signal(-1);
    __builtin_amdgcn_s_barrier_wait(-1);
    asm volatile("" ::: "memory");
#else
    __syncthreads();
#endif
}

// ---------------------------------------------------------------------------
// k_concat: all_emb = [user_table ; item_table]  (8192 x 64 f32)
// ---------------------------------------------------------------------------
__global__ void k_concat(const float* __restrict__ ut, const float* __restrict__ it,
                         float* __restrict__ emb) {
    int i = blockIdx.x * blockDim.x + threadIdx.x;
    if (i >= NN * D) return;
    emb[i] = (i < N_USERS * D) ? ut[i] : it[i - N_USERS * D];
}

// ---------------------------------------------------------------------------
// k_adj_pack: one-time adj f32 -> bf16 (0/1 exact). Halves the dominant HBM
// stream for all 4 subsequent passes. 8 elems / thread, b128 in / b128 out.
// ---------------------------------------------------------------------------
__global__ void k_adj_pack(const float* __restrict__ adj, unsigned short* __restrict__ abf) {
    size_t i = ((size_t)blockIdx.x * blockDim.x + threadIdx.x) * 8;
    if (i >= (size_t)NN * NN) return;
    float4 x0 = *(const float4*)(adj + i);
    float4 x1 = *(const float4*)(adj + i + 4);
    uint4 o;
    o.x = ((unsigned)f32_to_bf16(x0.y) << 16) | f32_to_bf16(x0.x);
    o.y = ((unsigned)f32_to_bf16(x0.w) << 16) | f32_to_bf16(x0.z);
    o.z = ((unsigned)f32_to_bf16(x1.y) << 16) | f32_to_bf16(x1.x);
    o.w = ((unsigned)f32_to_bf16(x1.w) << 16) | f32_to_bf16(x1.z);
    *(uint4*)(abf + i) = o;
}

// ---------------------------------------------------------------------------
// k_pack_b: pack an 8192x64 f32 matrix into the WMMA B layout (bf16).
// Slot index = ((kt*4 + nt)*32 + lane); each slot = 16 contiguous bf16:
// column n = nt*16 + lane%16, K = kt*32 + (lane/16)*8 + {0..7, 16..23}.
// One 4 KB K-tile = 2048 contiguous bf16.
// ---------------------------------------------------------------------------
__global__ void k_pack_b(const float* __restrict__ X, unsigned short* __restrict__ Bp) {
    int idx = blockIdx.x * blockDim.x + threadIdx.x;       // 256 kt * 4 nt * 32 lanes
    if (idx >= 256 * 4 * 32) return;
    int lane = idx & 31, nt = (idx >> 5) & 3, kt = idx >> 7;
    int n  = nt * 16 + (lane & 15);
    int k0 = kt * 32 + (lane >> 4) * 8;
    unsigned short* dst = Bp + (size_t)idx * 16;
#pragma unroll
    for (int i = 0; i < 16; ++i) dst[i] = f32_to_bf16(X[(size_t)(k0 + (i & 7) + (i >> 3) * 16) * D + n]);
}

// ---------------------------------------------------------------------------
// GCN aggregation GEMM, bf16-adj fast path:
//   out[8192x64] = adj_bf16[8192x8192] @ Xpacked
// A stream: per-lane register double buffer (HBM, loadcnt), prefetch kept in
//           flight across the barrier (raw split barrier, no loadcnt drain).
// B tiles : staged per-workgroup into double-buffered LDS via async copy
//           (ASYNCcnt), consumed via ds_load -> WMMA.
// One wave per 16-row strip; workgroup = 8 waves = 128 rows.
// ---------------------------------------------------------------------------
__global__ void __launch_bounds__(256) k_adj_gemm_bf(const unsigned short* __restrict__ abf,
                                                     const unsigned short* __restrict__ Bp,
                                                     float* __restrict__ out) {
    __shared__ unsigned short Bs[2][2048];                 // 2 x 4 KB K-tiles
    int tid  = threadIdx.x;
    int wave = blockIdx.x * 8 + (tid >> 5);                // 0..511
    int lane = tid & 31;
    int m    = lane & 15;
    int half = lane >> 4;
    const unsigned short* arow = abf + (size_t)(wave * 16 + m) * NN + half * 8;
    v8f acc0 = {}, acc1 = {}, acc2 = {}, acc3 = {};
    // prologue: stage B tile 0, load A tile 0
    stage_b16B(Bp + (size_t)tid * 8, &Bs[0][tid * 8]);
    v16bf A = load_a_bf16src(arow);
    stage_fence();
    for (int kt = 0; kt < 256; ++kt) {
        int buf = kt & 1;
        int ktn = (kt + 1 < 256) ? kt + 1 : 255;           // clamped (redundant tail)
        stage_b16B(Bp + (size_t)ktn * 2048 + tid * 8, &Bs[buf ^ 1][tid * 8]);
        v16bf An = load_a_bf16src(arow + ktn * 32);
        const v16u* ls = (const v16u*)&Bs[buf][0];
        v16bf B0 = __builtin_bit_cast(v16bf, ls[lane]);
        v16bf B1 = __builtin_bit_cast(v16bf, ls[lane + 32]);
        v16bf B2 = __builtin_bit_cast(v16bf, ls[lane + 64]);
        v16bf B3 = __builtin_bit_cast(v16bf, ls[lane + 96]);
        acc0 = wmma_bf16(A, B0, acc0);
        acc1 = wmma_bf16(A, B1, acc1);
        acc2 = wmma_bf16(A, B2, acc2);
        acc3 = wmma_bf16(A, B3, acc3);
        A = An;
        stage_fence();                                     // staged tile visible
    }
#pragma unroll
    for (int v = 0; v < 8; ++v) {
        float* orow = out + (size_t)(wave * 16 + half * 8 + v) * D + m;
        orow[0]  = acc0[v];
        orow[16] = acc1[v];
        orow[32] = acc2[v];
        orow[48] = acc3[v];
    }
}

// f32-adj fallback (used when ws is too small for the 128 MB bf16 adj copy)
__global__ void __launch_bounds__(256) k_adj_gemm_f32(const float* __restrict__ adj,
                                                      const unsigned short* __restrict__ Bp,
                                                      float* __restrict__ out) {
    __shared__ unsigned short Bs[2][2048];
    int tid  = threadIdx.x;
    int wave = blockIdx.x * 8 + (tid >> 5);
    int lane = tid & 31;
    int m    = lane & 15;
    int half = lane >> 4;
    const float* arow = adj + (size_t)(wave * 16 + m) * NN + half * 8;
    v8f acc0 = {}, acc1 = {}, acc2 = {}, acc3 = {};
    stage_b16B(Bp + (size_t)tid * 8, &Bs[0][tid * 8]);
    v16bf A = load_a_f32src(arow);
    stage_fence();
    for (int kt = 0; kt < 256; ++kt) {
        int buf = kt & 1;
        int ktn = (kt + 1 < 256) ? kt + 1 : 255;
        stage_b16B(Bp + (size_t)ktn * 2048 + tid * 8, &Bs[buf ^ 1][tid * 8]);
        v16bf An = load_a_f32src(arow + ktn * 32);
        const v16u* ls = (const v16u*)&Bs[buf][0];
        v16bf B0 = __builtin_bit_cast(v16bf, ls[lane]);
        v16bf B1 = __builtin_bit_cast(v16bf, ls[lane + 32]);
        v16bf B2 = __builtin_bit_cast(v16bf, ls[lane + 64]);
        v16bf B3 = __builtin_bit_cast(v16bf, ls[lane + 96]);
        acc0 = wmma_bf16(A, B0, acc0);
        acc1 = wmma_bf16(A, B1, acc1);
        acc2 = wmma_bf16(A, B2, acc2);
        acc3 = wmma_bf16(A, B3, acc3);
        A = An;
        stage_fence();
    }
#pragma unroll
    for (int v = 0; v < 8; ++v) {
        float* orow = out + (size_t)(wave * 16 + half * 8 + v) * D + m;
        orow[0]  = acc0[v];
        orow[16] = acc1[v];
        orow[32] = acc2[v];
        orow[48] = acc3[v];
    }
}

// ---------------------------------------------------------------------------
// Fused GAT attention (dense-softmax semantics, e/attn never materialized):
//   w_ij = exp( leaky(r_i + c_j + b) * adj_ij ),  adj in {0,1}
//   out_i = relu( (sum_j w_ij h_j) / (sum_j w_ij) )
// adj + c streams double-buffered in registers; H tiles staged in LDS.
// ---------------------------------------------------------------------------
__device__ __forceinline__ void load_adj16(const float* p, float* av) {
    float4 a0 = *(const float4*)(p);      float4 a1 = *(const float4*)(p + 4);
    float4 a2 = *(const float4*)(p + 16); float4 a3 = *(const float4*)(p + 20);
    av[0]=a0.x; av[1]=a0.y; av[2]=a0.z;  av[3]=a0.w;
    av[4]=a1.x; av[5]=a1.y; av[6]=a1.z;  av[7]=a1.w;
    av[8]=a2.x; av[9]=a2.y; av[10]=a2.z; av[11]=a2.w;
    av[12]=a3.x;av[13]=a3.y;av[14]=a3.z; av[15]=a3.w;
}
__device__ __forceinline__ void load_adj16(const unsigned short* p, float* av) {
    v8u lo = *(const v8u*)(p);
    v8u hi = *(const v8u*)(p + 16);
#pragma unroll
    for (int i = 0; i < 8; ++i) {
        av[i]     = __uint_as_float((unsigned)lo[i] << 16);   // bf16 0/1 exact
        av[8 + i] = __uint_as_float((unsigned)hi[i] << 16);
    }
}

template <typename AT>
__global__ void __launch_bounds__(256) k_attn(const AT* __restrict__ adjp,
                                              const float* __restrict__ rv,
                                              const float* __restrict__ cv,
                                              const float* __restrict__ attn_b,
                                              const unsigned short* __restrict__ Hp,
                                              float* __restrict__ out) {
    __shared__ unsigned short Bs[2][2048];
    int tid  = threadIdx.x;
    int wave = blockIdx.x * 8 + (tid >> 5);
    int lane = tid & 31;
    int m    = lane & 15;
    int half = lane >> 4;
    int row  = wave * 16 + m;
    float ri = rv[row];
    float ab = attn_b[0];
    const AT*    arow = adjp + (size_t)row * NN + half * 8;
    const float* crow = cv + half * 8;
    v8f acc0 = {}, acc1 = {}, acc2 = {}, acc3 = {};
    float zsum = 0.f;
    float av[16], cvv[16];
    stage_b16B(Hp + (size_t)tid * 8, &Bs[0][tid * 8]);
    load_adj16(arow, av);
    load_adj16(crow, cvv);
    stage_fence();
    for (int jt = 0; jt < 256; ++jt) {
        int buf = jt & 1;
        int jn  = (jt + 1 < 256) ? jt + 1 : 255;
        stage_b16B(Hp + (size_t)jn * 2048 + tid * 8, &Bs[buf ^ 1][tid * 8]);
        float avn[16], cvn[16];
        load_adj16(arow + jn * 32, avn);
        load_adj16(crow + jn * 32, cvn);
        v16u wu;
#pragma unroll
        for (int i = 0; i < 16; ++i) {
            float e  = ri + cvv[i] + ab;
            float le = (e >= 0.f) ? e : 0.01f * e;     // leaky relu
            float w  = __expf(le * av[i]);             // exp(0)=1 off-edge: dense softmax
            zsum += w;
            wu[i] = f32_to_bf16(w);
        }
        v16bf W = __builtin_bit_cast(v16bf, wu);
        const v16u* ls = (const v16u*)&Bs[buf][0];
        v16bf B0 = __builtin_bit_cast(v16bf, ls[lane]);
        v16bf B1 = __builtin_bit_cast(v16bf, ls[lane + 32]);
        v16bf B2 = __builtin_bit_cast(v16bf, ls[lane + 64]);
        v16bf B3 = __builtin_bit_cast(v16bf, ls[lane + 96]);
        acc0 = wmma_bf16(W, B0, acc0);
        acc1 = wmma_bf16(W, B1, acc1);
        acc2 = wmma_bf16(W, B2, acc2);
        acc3 = wmma_bf16(W, B3, acc3);
#pragma unroll
        for (int i = 0; i < 16; ++i) { av[i] = avn[i]; cvv[i] = cvn[i]; }
        stage_fence();
    }
    // Row sum: lanes m and m+16 cover complementary K chunks of row m.
    float ztot = zsum + __shfl_xor(zsum, 16, 32);
#pragma unroll
    for (int v = 0; v < 8; ++v) {
        int   r = half * 8 + v;
        float Z = __shfl(ztot, r, 32);
        float* orow = out + (size_t)(wave * 16 + r) * D + m;
        orow[0]  = fmaxf(acc0[v] / Z, 0.f);
        orow[16] = fmaxf(acc1[v] / Z, 0.f);
        orow[32] = fmaxf(acc2[v] / Z, 0.f);
        orow[48] = fmaxf(acc3[v] / Z, 0.f);
    }
}

// ---------------------------------------------------------------------------
// k_xw: out[Mx64] = (relu?)(A[Mx64] @ W[64x64] + b)   (small, cache-resident)
// ---------------------------------------------------------------------------
__global__ void k_xw(const float* __restrict__ A, const float* __restrict__ W,
                     const float* __restrict__ bias, float* __restrict__ out,
                     int M, int relu) {
    int idx = blockIdx.x * blockDim.x + threadIdx.x;
    if (idx >= M * D) return;
    int row = idx >> 6, col = idx & 63;
    const float* a = A + (size_t)row * D;
    float s = bias[col];
#pragma unroll 8
    for (int k = 0; k < D; ++k) s += a[k] * W[k * D + col];
    if (relu) s = fmaxf(s, 0.f);
    out[idx] = s;
}

// k_rc: r[i] = h_i . a1 ; c[i] = h_i . a2
__global__ void k_rc(const float* __restrict__ H, const float* __restrict__ aW,
                     float* __restrict__ r, float* __restrict__ c) {
    int row = blockIdx.x * blockDim.x + threadIdx.x;
    if (row >= NN) return;
    const float* h = H + (size_t)row * D;
    float s1 = 0.f, s2 = 0.f;
#pragma unroll 8
    for (int k = 0; k < D; ++k) { s1 += h[k] * aW[k]; s2 += h[k] * aW[D + k]; }
    r[row] = s1; c[row] = s2;
}

// k_lw: softmax of the 2 layer weights
__global__ void k_lw(const float* __restrict__ w, float* __restrict__ lw) {
    if (threadIdx.x == 0 && blockIdx.x == 0) {
        float m  = fmaxf(w[0], w[1]);
        float e0 = __expf(w[0] - m), e1 = __expf(w[1] - m);
        float inv = 1.f / (e0 + e1);
        lw[0] = e0 * inv; lw[1] = e1 * inv;
    }
}

// k_fuse: fused (+)= lw[li] * relu([gcn, gat] @ fuse_W + fuse_b)
__global__ void k_fuse(const float* __restrict__ gcn, const float* __restrict__ gat,
                       const float* __restrict__ fW, const float* __restrict__ fb,
                       const float* __restrict__ lw, int li,
                       float* __restrict__ fused, int accum) {
    int idx = blockIdx.x * blockDim.x + threadIdx.x;
    if (idx >= NN * D) return;
    int row = idx >> 6, col = idx & 63;
    const float* g1 = gcn + (size_t)row * D;
    const float* g2 = gat + (size_t)row * D;
    float s = fb[col];
#pragma unroll 8
    for (int k = 0; k < D; ++k) s += g1[k] * fW[k * D + col];
#pragma unroll 8
    for (int k = 0; k < D; ++k) s += g2[k] * fW[(D + k) * D + col];
    s = fmaxf(s, 0.f) * lw[li];
    fused[idx] = accum ? (fused[idx] + s) : s;
}

// ---------------------------------------------------------------------------
// k_head: gather, actor projection, scores, critic MLPs. One thread per sample.
// ---------------------------------------------------------------------------
__global__ void k_head(const float* __restrict__ fused,
                       const int* __restrict__ users, const int* __restrict__ pos,
                       const int* __restrict__ neg,
                       const float* __restrict__ aW, const float* __restrict__ ab,
                       const float* __restrict__ cW1, const float* __restrict__ cb1,
                       const float* __restrict__ cW2, const float* __restrict__ cb2,
                       float* __restrict__ out) {
    int b = blockIdx.x * blockDim.x + threadIdx.x;
    if (b >= BB) return;
    const float* u = fused + (size_t)users[b] * D;
    const float* p = fused + (size_t)(N_USERS + pos[b]) * D;
    const float* n = fused + (size_t)(N_USERS + neg[b]) * D;
    float ps = 0.f, ns = 0.f;
    for (int j = 0; j < D; ++j) {
        float s = ab[j];
#pragma unroll 8
        for (int k = 0; k < D; ++k) s += u[k] * aW[k * D + j];
        ps += s * p[j];
        ns += s * n[j];
    }
    out[b]      = ps;
    out[BB + b] = ns;
    float cp = cb2[0], cn = cb2[0];
    for (int j = 0; j < D; ++j) {
        float su = cb1[j];
#pragma unroll 8
        for (int k = 0; k < D; ++k) su += u[k] * cW1[k * D + j];
        float sp = su, sn = su;
#pragma unroll 8
        for (int k = 0; k < D; ++k) { float w = cW1[(D + k) * D + j]; sp += p[k] * w; sn += n[k] * w; }
        cp += fmaxf(sp, 0.f) * cW2[j];
        cn += fmaxf(sn, 0.f) * cW2[j];
    }
    out[2 * BB + b] = cp;
    out[3 * BB + b] = cn;
}

// ---------------------------------------------------------------------------
extern "C" void kernel_launch(void* const* d_in, const int* in_sizes, int n_in,
                              void* d_out, int out_size, void* d_ws, size_t ws_size,
                              hipStream_t stream) {
    const float* adj   = (const float*)d_in[0];
    const int*   users = (const int*)d_in[1];
    const int*   pos   = (const int*)d_in[2];
    const int*   neg   = (const int*)d_in[3];
    const float* utab  = (const float*)d_in[4];
    const float* itab  = (const float*)d_in[5];
    const float* gcnW  = (const float*)d_in[6];   // (2,64,64)
    const float* gcnB  = (const float*)d_in[7];   // (2,64)
    const float* gatW  = (const float*)d_in[8];
    const float* gatB  = (const float*)d_in[9];
    const float* attW  = (const float*)d_in[10];  // (2,128)
    const float* attB  = (const float*)d_in[11];  // (2,)
    const float* fuseW = (const float*)d_in[12];  // (2,128,64)
    const float* fuseB = (const float*)d_in[13];  // (2,64)
    const float* actW  = (const float*)d_in[14];
    const float* actB  = (const float*)d_in[15];
    const float* cW1   = (const float*)d_in[16];
    const float* cb1   = (const float*)d_in[17];
    const float* cW2   = (const float*)d_in[18];
    const float* cb2   = (const float*)d_in[19];
    const float* lwraw = (const float*)d_in[20];
    float* out = (float*)d_out;
    char*  ws  = (char*)d_ws;

    const size_t MAT = (size_t)NN * D * sizeof(float);      // 2 MB
    float* emb   = (float*)(ws + 0 * MAT);
    float* gcn1  = (float*)(ws + 1 * MAT);
    float* gcn2  = (float*)(ws + 2 * MAT);
    float* gat1  = (float*)(ws + 3 * MAT);
    float* gat2  = (float*)(ws + 4 * MAT);
    float* hbuf  = (float*)(ws + 5 * MAT);
    float* agg   = (float*)(ws + 6 * MAT);
    float* fused = (float*)(ws + 7 * MAT);
    unsigned short* pack = (unsigned short*)(ws + 8 * MAT); // 1 MB
    float* rv = (float*)(ws + 8 * MAT + (size_t)NN * D * 2);
    float* cv = rv + NN;
    float* lw = cv + NN;
    const size_t FIXED = 18ull * 1024 * 1024;               // everything above
    unsigned short* adjbf = (unsigned short*)(ws + FIXED);  // 128 MB (optional)
    const bool use_bf_adj = (ws_size >= FIXED + (size_t)NN * NN * 2);

    const int T = 256;
    k_concat<<<(NN * D + T - 1) / T, T, 0, stream>>>(utab, itab, emb);
    k_lw<<<1, 1, 0, stream>>>(lwraw, lw);
    if (use_bf_adj) {
        k_adj_pack<<<((size_t)NN * NN / 8 + T - 1) / T, T, 0, stream>>>(adj, adjbf);
    }

    const float* gcn_in[2]  = {emb, gcn1};
    float*       gcn_out[2] = {gcn1, gcn2};
    const float* gat_in[2]  = {emb, gat1};
    float*       gat_out[2] = {gat1, gat2};

    for (int l = 0; l < 2; ++l) {
        // --- GCN layer: agg = adj @ X ; gcn_out = relu(agg @ W + b) ---
        k_pack_b<<<(256 * 4 * 32 + T - 1) / T, T, 0, stream>>>(gcn_in[l], pack);
        if (use_bf_adj)
            k_adj_gemm_bf<<<64, 256, 0, stream>>>(adjbf, pack, agg);
        else
            k_adj_gemm_f32<<<64, 256, 0, stream>>>(adj, pack, agg);
        k_xw<<<(NN * D + T - 1) / T, T, 0, stream>>>(agg, gcnW + l * D * D, gcnB + l * D,
                                                     gcn_out[l], NN, 1);
        // --- GAT layer: h = X @ W + b ; r,c ; fused softmax-agg ---
        k_xw<<<(NN * D + T - 1) / T, T, 0, stream>>>(gat_in[l], gatW + l * D * D, gatB + l * D,
                                                     hbuf, NN, 0);
        k_rc<<<(NN + T - 1) / T, T, 0, stream>>>(hbuf, attW + l * 2 * D, rv, cv);
        k_pack_b<<<(256 * 4 * 32 + T - 1) / T, T, 0, stream>>>(hbuf, pack);
        if (use_bf_adj)
            k_attn<unsigned short><<<64, 256, 0, stream>>>(adjbf, rv, cv, attB + l, pack, gat_out[l]);
        else
            k_attn<float><<<64, 256, 0, stream>>>(adj, rv, cv, attB + l, pack, gat_out[l]);
    }

    // --- fuse layers ---
    k_fuse<<<(NN * D + T - 1) / T, T, 0, stream>>>(gcn1, gat1, fuseW + 0 * 2 * D * D,
                                                   fuseB + 0 * D, lw, 0, fused, 0);
    k_fuse<<<(NN * D + T - 1) / T, T, 0, stream>>>(gcn2, gat2, fuseW + 1 * 2 * D * D,
                                                   fuseB + 1 * D, lw, 1, fused, 1);

    // --- heads ---
    k_head<<<(BB + T - 1) / T, T, 0, stream>>>(fused, users, pos, neg, actW, actB,
                                               cW1, cb1, cW2, cb2, out);
}
